// VectorQuantizer_71949292143124
// MI455X (gfx1250) — matrix operand order repaired and checked by reading, not compiled, not run on hardware
//
#include <hip/hip_runtime.h>
#include <stdint.h>

typedef __attribute__((ext_vector_type(8)))  __bf16 v8bf;
typedef __attribute__((ext_vector_type(16))) __bf16 v16bf;
typedef __attribute__((ext_vector_type(8)))  float  v8f;

#define Ntok   16384          // B*S
#define Dd     256            // code dim
#define Kc     8192           // codebook size
#define MBLK   64             // rows per workgroup
#define NBLK   64             // codes per tile iteration
#define NT     (Kc / NBLK)    // 128 tile iterations
#define PITCH  264            // bf16 per LDS row (256 + 8 pad -> bank stride 4)

__device__ __forceinline__ unsigned int pk2bf(float a, float b) {
  unsigned int ua = __float_as_uint(a);
  unsigned int ub = __float_as_uint(b);
  ua = (ua + 0x7FFFu + ((ua >> 16) & 1u)) >> 16;   // round-to-nearest-even
  ub = (ub + 0x7FFFu + ((ub >> 16) & 1u)) >> 16;
  return (ub << 16) | (ua & 0xFFFFu);
}

// ---- Phase 1a: hidden fp32 -> bf16 (8 elems / thread) -------------------
__global__ __launch_bounds__(256)
void k_cvt_hidden(const float* __restrict__ h, uint32_t* __restrict__ abf) {
  int t = blockIdx.x * 256 + threadIdx.x;
  const float4* h4 = (const float4*)h;
  float4 x = h4[(size_t)t * 2 + 0];
  float4 y = h4[(size_t)t * 2 + 1];
  uint4 p;
  p.x = pk2bf(x.x, x.y); p.y = pk2bf(x.z, x.w);
  p.z = pk2bf(y.x, y.y); p.w = pk2bf(y.z, y.w);
  ((uint4*)abf)[t] = p;
}

// ---- Phase 1b: codebook fp32 -> bf16 + ||c||^2 (1 wave / row) -----------
__global__ __launch_bounds__(256)
void k_cvt_codebook(const float* __restrict__ cb, uint32_t* __restrict__ cbf,
                    float* __restrict__ cnorm) {
  int wave = threadIdx.x >> 5, lane = threadIdx.x & 31;
  int row  = blockIdx.x * 8 + wave;
  const float4* c4 = (const float4*)(cb + (size_t)row * Dd);
  float4 x = c4[lane * 2 + 0];
  float4 y = c4[lane * 2 + 1];
  uint4 p;
  p.x = pk2bf(x.x, x.y); p.y = pk2bf(x.z, x.w);
  p.z = pk2bf(y.x, y.y); p.w = pk2bf(y.z, y.w);
  ((uint4*)cbf)[(size_t)row * 32 + lane] = p;   // 256 bf16 = 32 uint4 per row
  float s = x.x*x.x + x.y*x.y + x.z*x.z + x.w*x.w
          + y.x*y.x + y.y*y.y + y.z*y.z + y.w*y.w;
  for (int off = 16; off > 0; off >>= 1) s += __shfl_xor(s, off, 32);
  if (lane == 0) cnorm[row] = s;
}

// ---- Phase 2: WMMA distance + running argmin ----------------------------
__global__ __launch_bounds__(256, 2)
void k_vq_argmin(const uint32_t* __restrict__ abf, const uint32_t* __restrict__ cbf,
                 const float* __restrict__ cnorm, int* __restrict__ outIdx) {
  __shared__ __bf16 As[MBLK * PITCH];
  __shared__ __bf16 Bs[2][NBLK * PITCH];
  __shared__ float sredv[MBLK];
  __shared__ int   sredi[MBLK];

  const int tid = threadIdx.x;
  const int rowBase = blockIdx.x * MBLK;
  const uint4* gA = (const uint4*)abf;
  const uint4* gC = (const uint4*)cbf;

  const int lr  = tid >> 2;   // loader row 0..63
  const int seg = tid & 3;    // 64 bf16 (8 uint4) per segment

  { // stage A block (64 rows x 256 bf16)
    const uint4* src = gA + (size_t)(rowBase + lr) * 32 + seg * 8;
    uint4* dst = (uint4*)&As[lr * PITCH + seg * 64];
    #pragma unroll
    for (int j = 0; j < 8; ++j) dst[j] = src[j];
  }
  { // stage codebook tile 0
    const uint4* src = gC + (size_t)lr * 32 + seg * 8;
    uint4* dst = (uint4*)&Bs[0][lr * PITCH + seg * 64];
    #pragma unroll
    for (int j = 0; j < 8; ++j) dst[j] = src[j];
  }
  __syncthreads();

  const int wave = tid >> 5, lane = tid & 31;
  const int rm = wave & 3;        // row tile within block
  const int cg = wave >> 2;       // col group (0/1): codes cg*32..cg*32+31
  const int laneRow = lane & 15;
  const int laneHi  = lane >> 4;

  float minv[8]; int mini[8];
  #pragma unroll
  for (int i = 0; i < 8; ++i) { minv[i] = __builtin_inff(); mini[i] = 0; }

  // A frag: lane holds row (rm*16+laneRow), 16 consecutive K at ks*32 + laneHi*16
  // (matching the B fragment K order). A is invariant across
  // the whole code sweep -> hoist all 8 fragments (full D=256) into VGPRs.
  v16bf afrag[8];
  {
    const __bf16* aBase = &As[(rm * 16 + laneRow) * PITCH + laneHi * 16];
    #pragma unroll
    for (int ks = 0; ks < 8; ++ks) {
      v8bf a0 = *(const v8bf*)(aBase + ks * 32);
      v8bf a1 = *(const v8bf*)(aBase + ks * 32 + 8);
      afrag[ks] = __builtin_shufflevector(a0, a1,
                    0,1,2,3,4,5,6,7,8,9,10,11,12,13,14,15);
    }
  }

  // LDS byte offsets of the two B buffers for this thread's loader slice.
  // Generic pointers to __shared__ live in the LDS aperture: addr[31:0] IS the
  // LDS byte address the async-load VDST operand wants.
  uint32_t ldsB[2];
  ldsB[0] = (uint32_t)(uintptr_t)&Bs[0][lr * PITCH + seg * 64];
  ldsB[1] = (uint32_t)(uintptr_t)&Bs[1][lr * PITCH + seg * 64];

  for (int kt = 0; kt < NT; ++kt) {
    const int cur = kt & 1;
    const int nxt = cur ^ 1;

    // Stage next codebook tile straight into LDS: async copy, no VGPR staging,
    // tracked by ASYNCcnt -> overlaps with the WMMA body below.
    if (kt + 1 < NT) {
      const char* src =
          (const char*)(gC + ((size_t)(kt + 1) * NBLK + lr) * 32 + seg * 8);
      uint32_t dst = ldsB[nxt];
      #pragma unroll
      for (int j = 0; j < 8; ++j) {
        asm volatile("global_load_async_to_lds_b128 %0, %1, off"
                     :: "v"(dst + (uint32_t)(j * 16)), "v"(src + j * 16)
                     : "memory");
      }
    }
    if (kt + 2 < NT)
      __builtin_prefetch(gC + ((size_t)(kt + 2) * NBLK + lr) * 32 + seg * 8, 0, 0);

    // B frag: lane holds code col (cg*32 + ct*16 + laneRow),
    // 16 consecutive K at ks*32 + laneHi*16 (same K order as the A fragment).
    v8f acc0 = {}; v8f acc1 = {};
    const __bf16* b0Base = &Bs[cur][(cg * 32 + laneRow) * PITCH + laneHi * 16];
    const __bf16* b1Base = b0Base + 16 * PITCH;
    #pragma unroll
    for (int ks = 0; ks < 8; ++ks) {
      v16bf b0 = *(const v16bf*)(b0Base + ks * 32);
      v16bf b1 = *(const v16bf*)(b1Base + ks * 32);
      acc0 = __builtin_amdgcn_wmma_f32_16x16x32_bf16(false, afrag[ks], false, b0,
                                                     (short)0, acc0, false, false);
      acc1 = __builtin_amdgcn_wmma_f32_16x16x32_bf16(false, afrag[ks], false, b1,
                                                     (short)0, acc1, false, false);
    }

    // dist = ||c||^2 - 2 x.c  (||x||^2 irrelevant for argmin)
    const int c0 = kt * NBLK + cg * 32 + laneRow;
    const int c1 = c0 + 16;
    float cn0 = cnorm[c0];
    float cn1 = cnorm[c1];
    #pragma unroll
    for (int i = 0; i < 8; ++i) {
      float d0 = cn0 - 2.0f * acc0[i];
      float d1 = cn1 - 2.0f * acc1[i];
      if (d0 < minv[i]) { minv[i] = d0; mini[i] = c0; }
      if (d1 < minv[i]) { minv[i] = d1; mini[i] = c1; }
    }

    // our async writes must have landed in LDS before the group-wide barrier
    asm volatile("s_wait_asynccnt 0x0" ::: "memory");
    __syncthreads();
  }

  // reduce over the 16 lanes holding different columns of the same rows
  #pragma unroll
  for (int i = 0; i < 8; ++i) {
    float v = minv[i]; int ix = mini[i];
    for (int off = 1; off < 16; off <<= 1) {
      float ov = __shfl_xor(v, off, 16);
      int   oi = __shfl_xor(ix, off, 16);
      if (ov < v || (ov == v && oi < ix)) { v = ov; ix = oi; }
    }
    minv[i] = v; mini[i] = ix;
  }
  // reduce across the two col-group waves sharing a row tile
  if (cg == 1 && laneRow == 0) {
    #pragma unroll
    for (int i = 0; i < 8; ++i) {
      int rib = rm * 16 + laneHi * 8 + i;
      sredv[rib] = minv[i]; sredi[rib] = mini[i];
    }
  }
  __syncthreads();
  if (cg == 0 && laneRow == 0) {
    #pragma unroll
    for (int i = 0; i < 8; ++i) {
      int rib = rm * 16 + laneHi * 8 + i;
      float ov = sredv[rib]; int oi = sredi[rib];
      float v = minv[i];     int ix = mini[i];
      if (ov < v || (ov == v && oi < ix)) { v = ov; ix = oi; }
      outIdx[rowBase + rib] = ix;
    }
  }
}

// ---- Phase 3: gather z_q (fp32), indices-as-float, loss partials --------
__global__ __launch_bounds__(256)
void k_gather_loss(const float* __restrict__ h, const float* __restrict__ cb,
                   const int* __restrict__ idx, float* __restrict__ out,
                   float* __restrict__ partials) {
  const int tid = threadIdx.x;
  const int rowBase = blockIdx.x * 64;
  const float4* h4 = (const float4*)h;
  const float4* c4 = (const float4*)cb;
  float4* o4 = (float4*)out;
  float lsum = 0.0f;
  for (int e = tid; e < 64 * 64; e += 256) {
    int r = e >> 6, cq = e & 63;
    int row = rowBase + r;
    int k = idx[row];
    float4 q = c4[(size_t)k * 64 + cq];
    float4 x = h4[(size_t)row * 64 + cq];
    o4[(size_t)row * 64 + cq] = q;
    float dx = x.x - q.x, dy = x.y - q.y, dz = x.z - q.z, dw = x.w - q.w;
    lsum += dx*dx + dy*dy + dz*dz + dw*dw;
  }
  float* outIdxF = out + (size_t)Ntok * Dd + 1;
  for (int r = tid; r < 64; r += 256)
    outIdxF[rowBase + r] = (float)idx[rowBase + r];
  int lane = tid & 31, wave = tid >> 5;
  for (int off = 16; off > 0; off >>= 1) lsum += __shfl_xor(lsum, off, 32);
  __shared__ float wsum[8];
  if (lane == 0) wsum[wave] = lsum;
  __syncthreads();
  if (tid == 0) {
    float s = 0.0f;
    for (int w = 0; w < 8; ++w) s += wsum[w];
    partials[blockIdx.x] = s;
  }
}

// ---- Phase 4: deterministic loss finalize -------------------------------
__global__ void k_finalize(const float* __restrict__ partials, float* __restrict__ out) {
  float s = 0.0f;
  for (int i = 0; i < 256; ++i) s += partials[i];
  // commitment_loss == codebook_loss numerically -> vq = 1.25 * mse
  out[(size_t)Ntok * Dd] = 1.25f * s / (float)((size_t)Ntok * Dd);
}

extern "C" void kernel_launch(void* const* d_in, const int* in_sizes, int n_in,
                              void* d_out, int out_size, void* d_ws, size_t ws_size,
                              hipStream_t stream) {
  const float* hidden   = (const float*)d_in[0];  // [4,4096,256]
  const float* codebook = (const float*)d_in[1];  // [8192,256]
  float* out = (float*)d_out;
  char* ws = (char*)d_ws;
  uint32_t* abf    = (uint32_t*)(ws);                // 8 MB   hidden bf16
  uint32_t* cbf    = (uint32_t*)(ws + 8388608);      // 4 MB   codebook bf16
  float*    cnorm  = (float*)   (ws + 12582912);     // 32 KB
  int*      idx    = (int*)     (ws + 12615680);     // 64 KB
  float*    parts  = (float*)   (ws + 12681216);     // 1 KB

  k_cvt_hidden  <<<2048, 256, 0, stream>>>(hidden, abf);
  k_cvt_codebook<<<1024, 256, 0, stream>>>(codebook, cbf, cnorm);
  k_vq_argmin   <<< 256, 256, 0, stream>>>(abf, cbf, cnorm, idx);
  k_gather_loss <<< 256, 256, 0, stream>>>(hidden, codebook, idx, out, parts);
  k_finalize    <<<   1,   1, 0, stream>>>(parts, out);
}
